// NeZhaSelfAttention_27934467293420
// MI455X (gfx1250) — matrix operand order repaired and checked by reading, not compiled, not run on hardware
//
#include <hip/hip_runtime.h>
#include <hip/hip_bf16.h>

// Problem constants (fixed by the harness)
#define BB  4
#define SS  1024
#define HH  1024
#define NHH 16
#define HDD 64
#define BHN 64      // BB*NHH
#define RV  256     // rel vocab padded 255 -> 256

typedef __attribute__((ext_vector_type(16))) __bf16 v16bf;
typedef __attribute__((ext_vector_type(8)))  __bf16 v8bf;
typedef __attribute__((ext_vector_type(16))) float  v16f;
typedef __attribute__((ext_vector_type(8)))  float  v8f;
typedef __attribute__((ext_vector_type(4)))  float  v4f;

#define WMMA_BF16(A, B, C) \
  __builtin_amdgcn_wmma_f32_16x16x32_bf16(false, (A), false, (B), (short)0, (C), false, false)

#define SHUF16(LO, HI) \
  __builtin_shufflevector((LO), (HI), 0, 1, 2, 3, 4, 5, 6, 7, 8, 9, 10, 11, 12, 13, 14, 15)

// ---- fragment loaders (wave32 layouts per CDNA5 ISA 7.12.2) -----------------
// A (16x32): lane m<16 holds K 0..7 (halves 0..7) and K 16..23 (halves 8..15);
// lane m+16 holds K 8..15 and K 24..31.  All loads are independent 16B chunks.
static __device__ inline v16bf a_frag_f32(const float* base, int ld, int lane) {
  int m = lane & 15;
  int ko = (lane & 16) ? 8 : 0;
  const float* p = base + (size_t)m * ld + ko;
  v4f a0 = *(const v4f*)(p);
  v4f a1 = *(const v4f*)(p + 4);
  v4f a2 = *(const v4f*)(p + 16);
  v4f a3 = *(const v4f*)(p + 20);
  v8f lo = __builtin_shufflevector(a0, a1, 0, 1, 2, 3, 4, 5, 6, 7);
  v8f hi = __builtin_shufflevector(a2, a3, 0, 1, 2, 3, 4, 5, 6, 7);
  v16f f = SHUF16(lo, hi);
  return __builtin_convertvector(f, v16bf);
}
static __device__ inline v16bf a_frag_bf16(const __bf16* base, int ld, int lane) {
  int m = lane & 15;
  int ko = (lane & 16) ? 8 : 0;
  const __bf16* p = base + (size_t)m * ld + ko;
  v8bf lo = *(const v8bf*)(p);
  v8bf hi = *(const v8bf*)(p + 16);
  return SHUF16(lo, hi);
}
// B (32x16): lane n<16 holds column n, K 0..7 and K 16..23; lane n+16 holds K 8..15 and K 24..31.
// Source layout: sb[n][k] (column n contiguous along K), row stride ld.
static __device__ inline v16bf b_frag_bf16(const __bf16* sb, int ld, int lane) {
  const __bf16* p = sb + (size_t)(lane & 15) * ld + ((lane & 16) ? 8 : 0);
  v8bf lo = *(const v8bf*)(p);
  v8bf hi = *(const v8bf*)(p + 16);
  return SHUF16(lo, hi);
}

// ---- kernel 0: bulk f32 -> bf16 conversion (8 elements / thread) ------------
__global__ void __launch_bounds__(256) nezha_cvt(const float* __restrict__ src,
                                                 __bf16* __restrict__ dst, int n8) {
  int t = blockIdx.x * 256 + threadIdx.x;
  if (t < n8) {
    const v4f* p = (const v4f*)src + 2 * (size_t)t;
    v4f x0 = p[0], x1 = p[1];
    v8f x = __builtin_shufflevector(x0, x1, 0, 1, 2, 3, 4, 5, 6, 7);
    *((v8bf*)dst + t) = __builtin_convertvector(x, v8bf);
  }
}

// ---- kernel 1: recover 255x64 sinusoidal table from rel_pos -----------------
// rel_pos[q,k,:] = table[clip(k-q,-127,127)+127, :]
__global__ void nezha_build_table(const float* __restrict__ rel,
                                  __bf16* __restrict__ tab,    // [256][64]
                                  __bf16* __restrict__ tabT) { // [64][256]
  int r = blockIdx.x;   // 0..255
  int d = threadIdx.x;  // 0..63
  float v = 0.f;
  if (r < 255) {
    int q = (r <= 127) ? (127 - r) : 0;
    int k = (r <= 127) ? 0 : (r - 127);
    v = rel[((size_t)q * SS + k) * HDD + d];
  }
  tab[r * HDD + d] = (__bf16)v;
  tabT[d * RV + r] = (__bf16)v;
}

// ---- kernel 2: fused QKV projection, bf16 WMMA (all-bf16 operands) ----------
// Y = X @ W^T + b ; Q pre-scaled by 1/sqrt(64); V stored transposed [bh][d][s].
__global__ void __launch_bounds__(256) nezha_qkv(
    const __bf16* __restrict__ Xb,
    const __bf16* __restrict__ Wqb, const float* __restrict__ bq,
    const __bf16* __restrict__ Wkb, const float* __restrict__ bk,
    const __bf16* __restrict__ Wvb, const float* __restrict__ bv,
    __bf16* __restrict__ Qb,   // [bh][s][d]
    __bf16* __restrict__ Kb,   // [bh][s][d]
    __bf16* __restrict__ Vt) { // [bh][d][s]
  int lane = threadIdx.x & 31;
  int w = blockIdx.x * 8 + (threadIdx.x >> 5);
  int mt = w >> 6, nt = w & 63;          // 256 M-tiles x 64 N-tiles
  int m0 = mt * 16, n0 = nt * 16;
  v8f cq = {}, ck = {}, cv = {};
#pragma unroll 2
  for (int kk = 0; kk < HH; kk += 32) {
    v16bf a  = a_frag_bf16(Xb  + (size_t)m0 * HH + kk, HH, lane);
    v16bf fq = b_frag_bf16(Wqb + (size_t)n0 * HH + kk, HH, lane);
    v16bf fk = b_frag_bf16(Wkb + (size_t)n0 * HH + kk, HH, lane);
    v16bf fv = b_frag_bf16(Wvb + (size_t)n0 * HH + kk, HH, lane);
    cq = WMMA_BF16(a, fq, cq);
    ck = WMMA_BF16(a, fk, ck);
    cv = WMMA_BF16(a, fv, cv);
  }
  int n = n0 + (lane & 15);
  float biq = bq[n], bik = bk[n], biv = bv[n];
  int nh = n >> 6, d = n & 63;
#pragma unroll
  for (int i = 0; i < 8; ++i) {
    int m = m0 + i + ((lane & 16) ? 8 : 0);
    int b = m >> 10, s = m & 1023;
    int bh = b * NHH + nh;
    Qb[((size_t)bh * SS + s) * HDD + d] = (__bf16)((cq[i] + biq) * 0.125f);
    Kb[((size_t)bh * SS + s) * HDD + d] = (__bf16)(ck[i] + bik);
    Vt[((size_t)bh * HDD + d) * SS + s] = (__bf16)(cv[i] + biv);
  }
}

// ---- kernel 3: QT = Qscaled @ table^T  [65536 x 256, K=64] ------------------
__global__ void __launch_bounds__(256) nezha_qt(
    const __bf16* __restrict__ Qb, const __bf16* __restrict__ tab,
    __bf16* __restrict__ QT) {     // [bh*S][256]
  int lane = threadIdx.x & 31;
  int w = blockIdx.x * 8 + (threadIdx.x >> 5);
  int mt = w >> 4, nt = w & 15;    // 4096 M-tiles x 16 N-tiles
  int m0 = mt * 16, r0 = nt * 16;
  v8f c = {};
#pragma unroll
  for (int kk = 0; kk < HDD; kk += 32) {
    v16bf a  = a_frag_bf16(Qb  + (size_t)m0 * HDD + kk, HDD, lane);
    v16bf bt = b_frag_bf16(tab + (size_t)r0 * HDD + kk, HDD, lane);
    c = WMMA_BF16(a, bt, c);
  }
  int r = r0 + (lane & 15);
#pragma unroll
  for (int i = 0; i < 8; ++i) {
    int m = m0 + i + ((lane & 16) ? 8 : 0);
    QT[(size_t)m * RV + r] = (__bf16)c[i];
  }
}

// ---- kernel 4: two-pass flash attention with relative-position terms --------
// grid = bh(64) x qblock(16); block = 128 (4 independent waves, 16 queries each)
__global__ void __launch_bounds__(128) nezha_flash(
    const __bf16* __restrict__ Qb, const __bf16* __restrict__ Kb,
    const __bf16* __restrict__ Vt, const __bf16* __restrict__ QT,
    const __bf16* __restrict__ tabT, float* __restrict__ out) {
  extern __shared__ char smem[];
  int lane = threadIdx.x & 31;
  int wid  = threadIdx.x >> 5;
  int bh = blockIdx.x >> 4;
  int qb = blockIdx.x & 15;
  int q0 = qb * 64 + wid * 16;

  float*  PS = (float*)smem + (size_t)wid * 16 * RV;              // 16KB/wave
  __bf16* PT = (__bf16*)(smem + 4 * 16 * RV * sizeof(float)) + wid * 16 * 40;

  const __bf16* Qbh = Qb + (size_t)bh * SS * HDD;
  const __bf16* Kbh = Kb + (size_t)bh * SS * HDD;
  const __bf16* Vbh = Vt + (size_t)bh * HDD * SS;
  const __bf16* QTb = QT + (size_t)bh * SS * RV;

  v16bf aq0 = a_frag_bf16(Qbh + (size_t)q0 * HDD +  0, HDD, lane);
  v16bf aq1 = a_frag_bf16(Qbh + (size_t)q0 * HDD + 32, HDD, lane);

  int qro  = (lane & 16) ? 8 : 0;  // row offset within tile for this half-wave
  int kcol = lane & 15;

  float mrow[8], lrow[8];
#pragma unroll
  for (int i = 0; i < 8; ++i) { mrow[i] = -1e30f; lrow[i] = 0.f; }

  // ---- pass 1: row max + exp-sum (no accumulator rescale needed later) ----
  for (int kk = 0; kk < SS; kk += 32) {
    __builtin_prefetch(Kbh + (size_t)(kk + 32) * HDD, 0, 1);
    v16bf b0 = b_frag_bf16(Kbh + (size_t)kk * HDD +  0, HDD, lane);
    v16bf b1 = b_frag_bf16(Kbh + (size_t)kk * HDD + 32, HDD, lane);
    v16bf b2 = b_frag_bf16(Kbh + (size_t)(kk + 16) * HDD +  0, HDD, lane);
    v16bf b3 = b_frag_bf16(Kbh + (size_t)(kk + 16) * HDD + 32, HDD, lane);
    v8f c0 = {}, c1 = {};
    c0 = WMMA_BF16(aq0, b0, c0);  c0 = WMMA_BF16(aq1, b1, c0);
    c1 = WMMA_BF16(aq0, b2, c1);  c1 = WMMA_BF16(aq1, b3, c1);
#pragma unroll
    for (int i = 0; i < 8; ++i) {
      int q = q0 + i + qro;
      int i0 = min(max(kk + kcol      - q, -127), 127) + 127;
      int i1 = min(max(kk + 16 + kcol - q, -127), 127) + 127;
      c0[i] += (float)QTb[(size_t)q * RV + i0];
      c1[i] += (float)QTb[(size_t)q * RV + i1];
      float v = fmaxf(c0[i], c1[i]);
#pragma unroll
      for (int o = 8; o > 0; o >>= 1) v = fmaxf(v, __shfl_xor(v, o, 16));
      float mn = fmaxf(mrow[i], v);
      float e  = __expf(c0[i] - mn) + __expf(c1[i] - mn);
#pragma unroll
      for (int o = 8; o > 0; o >>= 1) e += __shfl_xor(e, o, 16);
      lrow[i] = lrow[i] * __expf(mrow[i] - mn) + e;
      mrow[i] = mn;
    }
  }
  float rinv[8];
#pragma unroll
  for (int i = 0; i < 8; ++i) rinv[i] = 1.f / lrow[i];

  // zero this wave's PS histogram (wave-private; DS is in-order per wave)
  for (int t = lane; t < 16 * RV; t += 32) PS[t] = 0.f;

  // ---- pass 2: P = exp(s-m)/l ; ctx += P@V ; PS bins += P ----
  v8f acc[4] = {{}, {}, {}, {}};
  for (int kk = 0; kk < SS; kk += 32) {
    __builtin_prefetch(Vbh + kk + 32, 0, 1);
    v16bf b0 = b_frag_bf16(Kbh + (size_t)kk * HDD +  0, HDD, lane);
    v16bf b1 = b_frag_bf16(Kbh + (size_t)kk * HDD + 32, HDD, lane);
    v16bf b2 = b_frag_bf16(Kbh + (size_t)(kk + 16) * HDD +  0, HDD, lane);
    v16bf b3 = b_frag_bf16(Kbh + (size_t)(kk + 16) * HDD + 32, HDD, lane);
    v8f c0 = {}, c1 = {};
    c0 = WMMA_BF16(aq0, b0, c0);  c0 = WMMA_BF16(aq1, b1, c0);
    c1 = WMMA_BF16(aq0, b2, c1);  c1 = WMMA_BF16(aq1, b3, c1);
#pragma unroll
    for (int i = 0; i < 8; ++i) {
      int row = i + qro;
      int q   = q0 + row;
      int i0 = min(max(kk + kcol      - q, -127), 127) + 127;
      int i1 = min(max(kk + 16 + kcol - q, -127), 127) + 127;
      float p0 = __expf(c0[i] + (float)QTb[(size_t)q * RV + i0] - mrow[i]) * rinv[i];
      float p1 = __expf(c1[i] + (float)QTb[(size_t)q * RV + i1] - mrow[i]) * rinv[i];
      PT[row * 40 + kcol]      = (__bf16)p0;   // C-layout -> LDS
      PT[row * 40 + 16 + kcol] = (__bf16)p1;
      atomicAdd(&PS[row * RV + i0], p0);       // ds_add_f32 histogram
      atomicAdd(&PS[row * RV + i1], p1);
    }
    v16bf ap = a_frag_bf16(PT, 40, lane);      // LDS -> A-layout [16 x 32]
#pragma unroll
    for (int dt = 0; dt < 4; ++dt) {
      v16bf bv = b_frag_bf16(Vbh + (size_t)(dt * 16) * SS + kk, SS, lane);
      acc[dt] = WMMA_BF16(ap, bv, acc[dt]);
    }
  }

  // ---- pass 3: ctx += PS @ table  (K = 256) ----
  for (int rk = 0; rk < RV; rk += 32) {
    v16bf aps = a_frag_f32(PS + rk, RV, lane);
#pragma unroll
    for (int dt = 0; dt < 4; ++dt) {
      v16bf bt = b_frag_bf16(tabT + (size_t)(dt * 16) * RV + rk, RV, lane);
      acc[dt] = WMMA_BF16(aps, bt, acc[dt]);
    }
  }

  // ---- store ctx to out[b][q][nh*64+d] ----
  int b = bh >> 4, nh = bh & 15;
#pragma unroll
  for (int dt = 0; dt < 4; ++dt)
#pragma unroll
    for (int i = 0; i < 8; ++i) {
      int q = q0 + i + qro;
      int d = dt * 16 + kcol;
      out[((size_t)b * SS + q) * HH + nh * HDD + d] = acc[dt][i];
    }
}

extern "C" void kernel_launch(void* const* d_in, const int* in_sizes, int n_in,
                              void* d_out, int out_size, void* d_ws, size_t ws_size,
                              hipStream_t stream) {
  (void)in_sizes; (void)n_in; (void)out_size; (void)ws_size;
  const float* X   = (const float*)d_in[0];
  const float* Wq  = (const float*)d_in[1];
  const float* bq  = (const float*)d_in[2];
  const float* Wk  = (const float*)d_in[3];
  const float* bk  = (const float*)d_in[4];
  const float* Wv  = (const float*)d_in[5];
  const float* bv  = (const float*)d_in[6];
  const float* rel = (const float*)d_in[7];
  float* out = (float*)d_out;

  char* ws = (char*)d_ws;
  __bf16* Qb   = (__bf16*)(ws);                        //  8 MB [bh][s][d]
  __bf16* Kb   = (__bf16*)(ws + (size_t)8  * 1048576); //  8 MB [bh][s][d]
  __bf16* Vt   = (__bf16*)(ws + (size_t)16 * 1048576); //  8 MB [bh][d][s]
  __bf16* QT   = (__bf16*)(ws + (size_t)24 * 1048576); // 32 MB [bh*s][256]
  __bf16* tab  = (__bf16*)(ws + (size_t)56 * 1048576); // 32 KB [256][64]
  __bf16* tabT = (__bf16*)(ws + (size_t)56 * 1048576 + 65536); // 32 KB [64][256]
  __bf16* Xb   = (__bf16*)(ws + (size_t)57 * 1048576); //  8 MB [4096][1024]
  __bf16* Wqb  = (__bf16*)(ws + (size_t)65 * 1048576); //  2 MB
  __bf16* Wkb  = (__bf16*)(ws + (size_t)67 * 1048576); //  2 MB
  __bf16* Wvb  = (__bf16*)(ws + (size_t)69 * 1048576); //  2 MB

  nezha_build_table<<<256, 64, 0, stream>>>(rel, tab, tabT);
  nezha_cvt<<<2048, 256, 0, stream>>>(X,  Xb,  (BB * SS * HH) / 8);
  nezha_cvt<<<512,  256, 0, stream>>>(Wq, Wqb, (HH * HH) / 8);
  nezha_cvt<<<512,  256, 0, stream>>>(Wk, Wkb, (HH * HH) / 8);
  nezha_cvt<<<512,  256, 0, stream>>>(Wv, Wvb, (HH * HH) / 8);
  nezha_qkv<<<2048, 256, 0, stream>>>(Xb, Wqb, bq, Wkb, bk, Wvb, bv, Qb, Kb, Vt);
  nezha_qt<<<8192, 256, 0, stream>>>(Qb, tab, QT);
  size_t lds = (size_t)4 * 16 * RV * sizeof(float) + (size_t)4 * 16 * 40 * sizeof(__bf16);
  nezha_flash<<<64 * 16, 128, lds, stream>>>(Qb, Kb, Vt, QT, tabT, out);
}
